// LLMBinaryMultitaskMLPGenerator_42631845380886
// MI455X (gfx1250) — matrix operand order repaired and checked by reading, not compiled, not run on hardware
//
#include <hip/hip_runtime.h>
#include <hip/hip_bf16.h>
#include <math.h>

#define T_  16
#define D_  1024
#define H1_ 512
#define H2_ 256
#define BS_ 8192          // B*S = 4*2048
#define MTILE 32          // rows per workgroup

#define SX_LD  136        // padded row strides: (bytes/4)%64==4 -> conflict-free b128 loads
#define SH1_LD 520
#define SH2_LD 264

typedef __attribute__((ext_vector_type(16))) __bf16 v16bf;
typedef __attribute__((ext_vector_type(8)))  float  v8f;

union Frag {
    uint4 u[2];
    v16bf v;
};

__device__ __forceinline__ __bf16 to_bf16(float f) { return (__bf16)f; }

__device__ __forceinline__ v8f wmma_bf16(v16bf a, v16bf b, v8f c) {
    // D = A(16x32 bf16) * B(32x16 bf16) + C(16x16 f32)
    return __builtin_amdgcn_wmma_f32_16x16x32_bf16(
        /*neg_a=*/false, a, /*neg_b=*/false, b,
        /*c_mod=*/(short)0, c, /*reuse_a=*/false, /*reuse_b=*/false);
}

// A fragment: 16x32 bf16 tile from row-major LDS buffer.
// lanes 0-15: M=lane, K {0..7,16..23}; lanes 16-31: M=lane-16, K {8..15,24..31}
__device__ __forceinline__ v16bf load_a(const __bf16* base, int ldb, int m0, int kk, int lane) {
    const int row = m0 + (lane & 15);
    const int kh  = kk + ((lane >> 4) << 3);
    const uint4* p = reinterpret_cast<const uint4*>(base + row * ldb + kh);
    Frag f;
    f.u[0] = p[0];        // K = kh .. kh+7
    f.u[1] = p[2];        // K = kh+16 .. kh+23  (+32 bytes)
    return f.v;
}

// B fragment: 32x16 bf16 tile from [n][k] (K-contiguous) global weights.
// lanes 0-15: N=lane, K {0..7,16..23}; lanes 16-31: N=lane-16, K {8..15,24..31}
__device__ __forceinline__ Frag load_b(const __bf16* wt, int ldk, int n0, int k0, int lane) {
    const int n  = n0 + (lane & 15);
    const int kh = k0 + ((lane >> 4) << 3);
    const uint4* q = reinterpret_cast<const uint4*>(wt + (size_t)n * ldk + kh);
    Frag f;
    f.u[0] = q[0];
    f.u[1] = q[2];
    return f;
}

// ---------------- weight conversion prologues (fp32 -> bf16, [t][k][n] -> [t][n][k]) --------
__global__ void cvt_w1_kernel(const float* __restrict__ W1, __bf16* __restrict__ W1T) {
    const int e = blockIdx.x * 256 + threadIdx.x;     // T_*H1_*D_ elements
    const int k = e & (D_ - 1);
    const int n = (e >> 10) & (H1_ - 1);
    const int t = e >> 19;
    W1T[e] = to_bf16(W1[((size_t)(t * D_ + k)) * H1_ + n]);
}

__global__ void cvt_w2_kernel(const float* __restrict__ W2, __bf16* __restrict__ W2T) {
    const int e = blockIdx.x * 256 + threadIdx.x;     // T_*H2_*H1_ elements
    const int k = e & (H1_ - 1);
    const int n = (e >> 9) & (H2_ - 1);
    const int t = e >> 17;
    W2T[e] = to_bf16(W2[((size_t)(t * H1_ + k)) * H2_ + n]);
}

// ---------------- fused per-task MLP --------------------------------------------------------
// grid: (BS_/MTILE) * T_ blocks of 256 threads (8 waves).
__global__ void __launch_bounds__(256)
mlp_fused_kernel(const float* __restrict__ x,
                 const float* __restrict__ b1,
                 const float* __restrict__ b2,
                 const float* __restrict__ W3,
                 const float* __restrict__ b3,
                 const __bf16* __restrict__ w1t,   // [T][H1][D] bf16
                 const __bf16* __restrict__ w2t,   // [T][H2][H1] bf16
                 float* __restrict__ out)          // [BS_][T_]
{
    // Typed LDS (keeps addrspace(3) inference -> ds_load_b128, not flat/global):
    //   sx double buffer (2 x 32x136 bf16, dead after layer 1) overlaps sh2 via union.
    union XS2 {
        __bf16 sx[2][MTILE][SX_LD];       // 17408 B
        __bf16 sh2[MTILE][SH2_LD];        // 16896 B
    };
    __shared__ __align__(16) XS2   u;
    __shared__ __align__(16) __bf16 sh1[MTILE][SH1_LD];   // 33280 B
    __shared__ float sred[MTILE][8];                      //  1024 B

    const int tid  = threadIdx.x;
    const int lane = tid & 31;
    const int wv   = tid >> 5;            // 0..7
    const int t    = blockIdx.x & (T_ - 1);
    const int m_base = (blockIdx.x >> 4) * MTILE;

    const __bf16* w1t_t = w1t + (size_t)t * H1_ * D_;
    const __bf16* w2t_t = w2t + (size_t)t * H2_ * H1_;

    // per-thread x staging coordinates (4 float4 per thread per 32x128 chunk)
    const int sr[4]  = { (tid + 0) >> 5, (tid + 256) >> 5, (tid + 512) >> 5, (tid + 768) >> 5 };
    const int sc4    = (tid & 31) << 2;

    float4 xr[4];
    auto load_xchunk = [&](int k0) {
        #pragma unroll
        for (int i = 0; i < 4; ++i)
            xr[i] = *reinterpret_cast<const float4*>(
                x + ((size_t)(m_base + sr[i]) * T_ + t) * D_ + k0 + sc4);
    };
    auto store_xchunk = [&](int buf) {
        #pragma unroll
        for (int i = 0; i < 4; ++i) {
            __bf16* dst = &u.sx[buf][sr[i]][sc4];
            dst[0] = to_bf16(fmaxf(xr[i].x, 0.0f));
            dst[1] = to_bf16(fmaxf(xr[i].y, 0.0f));
            dst[2] = to_bf16(fmaxf(xr[i].z, 0.0f));
            dst[3] = to_bf16(fmaxf(xr[i].w, 0.0f));
        }
    };

    // ---------------- layer 1: H1[32x512] = relu(relu(X)[32x1024] @ W1 + b1) ---------------
    const int n1_base = wv * 64;          // this wave's 64 columns
    v8f acc1[2][4];
    #pragma unroll
    for (int mt = 0; mt < 2; ++mt)
        #pragma unroll
        for (int nt = 0; nt < 4; ++nt)
            #pragma unroll
            for (int i = 0; i < 8; ++i) acc1[mt][nt][i] = 0.0f;

    // prologue: stage chunk 0, prefetch B for k-step 0, warm WGP$ for layer-2 weights
    load_xchunk(0);
    Frag bbuf[2][4];
    #pragma unroll
    for (int nt = 0; nt < 4; ++nt)
        bbuf[0][nt] = load_b(w1t_t, D_, n1_base + nt * 16, 0, lane);
    __builtin_prefetch(w2t_t + (size_t)(wv * 32 + (lane & 31)) * H1_, 0, 0);
    store_xchunk(0);
    __syncthreads();

    for (int c = 0; c < 8; ++c) {         // 8 chunks of K=128 (rolled: lower VGPR pressure)
        const int buf = c & 1;
        if (c < 7) load_xchunk((c + 1) * 128);     // overlap next x chunk with WMMAs

        #pragma unroll
        for (int s = 0; s < 4; ++s) {              // K-steps of 32 within chunk
            const int p = s & 1;                   // (c*4+s)&1 == s&1 (c*4 even)
            const int knext = c * 128 + s * 32 + 32;
            if (knext < D_) {                      // prefetch B one k-step ahead
                #pragma unroll
                for (int nt = 0; nt < 4; ++nt)
                    bbuf[p ^ 1][nt] = load_b(w1t_t, D_, n1_base + nt * 16, knext, lane);
            }
            const v16bf a0 = load_a(&u.sx[buf][0][0], SX_LD, 0,  s * 32, lane);
            const v16bf a1 = load_a(&u.sx[buf][0][0], SX_LD, 16, s * 32, lane);
            #pragma unroll
            for (int nt = 0; nt < 4; ++nt) {
                acc1[0][nt] = wmma_bf16(a0, bbuf[p][nt].v, acc1[0][nt]);
                acc1[1][nt] = wmma_bf16(a1, bbuf[p][nt].v, acc1[1][nt]);
            }
        }
        if (c < 7) store_xchunk(buf ^ 1);          // safe: last readers of that buffer were
        __syncthreads();                           // in chunk c-1, one barrier behind
    }

    // epilogue: + b1, relu, store bf16 to LDS
    #pragma unroll
    for (int nt = 0; nt < 4; ++nt) {
        const int col  = n1_base + nt * 16 + (lane & 15);
        const float bias = b1[t * H1_ + col];
        #pragma unroll
        for (int mt = 0; mt < 2; ++mt)
            #pragma unroll
            for (int i = 0; i < 8; ++i) {
                const int row = mt * 16 + i + ((lane >> 4) << 3);
                sh1[row][col] = to_bf16(fmaxf(acc1[mt][nt][i] + bias, 0.0f));
            }
    }
    __syncthreads();

    // ---------------- layer 2: H2[32x256] = relu(H1 @ W2 + b2) -----------------------------
    const int n2_base = wv * 32;          // this wave's 32 columns
    v8f acc2[2][2];
    #pragma unroll
    for (int mt = 0; mt < 2; ++mt)
        #pragma unroll
        for (int nt = 0; nt < 2; ++nt)
            #pragma unroll
            for (int i = 0; i < 8; ++i) acc2[mt][nt][i] = 0.0f;

    Frag bb2[2][2];
    #pragma unroll
    for (int nt = 0; nt < 2; ++nt)
        bb2[0][nt] = load_b(w2t_t, H1_, n2_base + nt * 16, 0, lane);

    #pragma unroll
    for (int s = 0; s < 16; ++s) {        // K-steps of 32, B pipelined one step ahead
        const int p = s & 1;
        if (s < 15) {
            #pragma unroll
            for (int nt = 0; nt < 2; ++nt)
                bb2[p ^ 1][nt] = load_b(w2t_t, H1_, n2_base + nt * 16, (s + 1) * 32, lane);
        }
        const v16bf a0 = load_a(&sh1[0][0], SH1_LD, 0,  s * 32, lane);
        const v16bf a1 = load_a(&sh1[0][0], SH1_LD, 16, s * 32, lane);
        #pragma unroll
        for (int nt = 0; nt < 2; ++nt) {
            acc2[0][nt] = wmma_bf16(a0, bb2[p][nt].v, acc2[0][nt]);
            acc2[1][nt] = wmma_bf16(a1, bb2[p][nt].v, acc2[1][nt]);
        }
    }
    // sh2 aliases the sx buffers; their last readers are two barriers back -> safe to write.

    #pragma unroll
    for (int nt = 0; nt < 2; ++nt) {
        const int col  = n2_base + nt * 16 + (lane & 15);
        const float bias = b2[t * H2_ + col];
        #pragma unroll
        for (int mt = 0; mt < 2; ++mt)
            #pragma unroll
            for (int i = 0; i < 8; ++i) {
                const int row = mt * 16 + i + ((lane >> 4) << 3);
                u.sh2[row][col] = to_bf16(fmaxf(acc2[mt][nt][i] + bias, 0.0f));
            }
    }
    __syncthreads();

    // ---------------- layer 3: out[32] = sigmoid(H2 @ W3 + b3) -----------------------------
    {
        const int r = tid >> 3;           // row 0..31
        const int g = tid & 7;            // 8-way split of K=256
        const float* w3_t = W3 + t * H2_;
        float s = 0.0f;
        #pragma unroll
        for (int k = 0; k < 32; ++k) {
            const int kk = g * 32 + k;
            s += (float)u.sh2[r][kk] * w3_t[kk];
        }
        sred[r][g] = s;
    }
    __syncthreads();

    if (tid < MTILE) {
        float s = b3[t];
        #pragma unroll
        for (int g = 0; g < 8; ++g) s += sred[tid][g];
        const float o = 1.0f / (1.0f + __expf(-s));
        out[(size_t)(m_base + tid) * T_ + t] = o;
    }
}

// -------------------------------------------------------------------------------------------
extern "C" void kernel_launch(void* const* d_in, const int* in_sizes, int n_in,
                              void* d_out, int out_size, void* d_ws, size_t ws_size,
                              hipStream_t stream) {
    const float* x  = (const float*)d_in[0];
    const float* W1 = (const float*)d_in[1];
    const float* b1 = (const float*)d_in[2];
    const float* W2 = (const float*)d_in[3];
    const float* b2 = (const float*)d_in[4];
    const float* W3 = (const float*)d_in[5];
    const float* b3 = (const float*)d_in[6];
    float* out = (float*)d_out;

    // workspace: bf16 transposed weights
    __bf16* w1t = (__bf16*)d_ws;                                        // T*H1*D  = 16.78 MB
    __bf16* w2t = (__bf16*)((char*)d_ws + (size_t)T_ * H1_ * D_ * 2);   // T*H2*H1 =  4.19 MB

    cvt_w1_kernel<<<(T_ * H1_ * D_) / 256, 256, 0, stream>>>(W1, w1t);
    cvt_w2_kernel<<<(T_ * H2_ * H1_) / 256, 256, 0, stream>>>(W2, w2t);

    const int nblocks = (BS_ / MTILE) * T_;   // 256 * 16 = 4096
    mlp_fused_kernel<<<nblocks, 256, 0, stream>>>(x, b1, b2, W3, b3, w1t, w2t, out);
}
